// LoraQuantLinear_45621142618735
// MI455X (gfx1250) — matrix operand order repaired and checked by reading, not compiled, not run on hardware
//
#include <hip/hip_runtime.h>

// Problem sizes (fixed by the reference)
#define TOK  64
#define IN   4096
#define OUTD 14336
#define RNK  16
#define NBLK 64   // NF4 quant block

typedef _Float16 v16h __attribute__((ext_vector_type(16)));
typedef float    v8f  __attribute__((ext_vector_type(8)));
typedef int      v4i  __attribute__((ext_vector_type(4)));
typedef float    v4f  __attribute__((ext_vector_type(4)));

__constant__ float NF4_C[16] = {
    -1.0f, -0.6961928009986877f, -0.5250730514526367f, -0.39491748809814453f,
    -0.28444138169288635f, -0.18477343022823334f, -0.09105003625154495f, 0.0f,
    0.07958029955625534f, 0.16093020141124725f, 0.24611230194568634f, 0.33791524171829224f,
    0.44070982933044434f, 0.5626170039176941f, 0.7229568362236023f, 1.0f};

// NF4 lookup via cross-lane crossbar: lanes 0-15 hold the codebook in `tab`,
// ds_bpermute gathers entry `code` (byte index = code*4, bits[6:2] -> lane).
static __device__ __forceinline__ float nf4v(float tab, int code) {
    return __builtin_bit_cast(float,
        __builtin_amdgcn_ds_bpermute(code << 2, __builtin_bit_cast(int, tab)));
}
// v_cvt_pk_rtz_f16_f32: pack two f32 -> dword (lo = first element, matches B layout)
static __device__ __forceinline__ unsigned pk2(float lo, float hi) {
    return __builtin_bit_cast(unsigned, __builtin_amdgcn_cvt_pkrtz(lo, hi));
}

union U16h {
    v4i      q[2];
    unsigned u[8];
    v16h     h;
};

// ---------------- prologue 1: x f32 -> f16 into workspace ----------------
__global__ void cvt_x_kernel(const float* __restrict__ x, _Float16* __restrict__ xh) {
    int i = (blockIdx.x * blockDim.x + threadIdx.x) * 4;
    v4f v = *(const v4f*)(x + i);
    xh[i + 0] = (_Float16)v.x;
    xh[i + 1] = (_Float16)v.y;
    xh[i + 2] = (_Float16)v.z;
    xh[i + 3] = (_Float16)v.w;
}

// ------------- prologue 2: xa[t,r] = 2 * dot(x[t,:], A[r,:]) (f16) -------------
__global__ void xa_kernel(const float* __restrict__ x, const float* __restrict__ loraA,
                          _Float16* __restrict__ xa) {
    int gid  = blockIdx.x * blockDim.x + threadIdx.x;
    int wv   = gid >> 5;          // 1024 waves = 64 tokens * 16 ranks
    int lane = threadIdx.x & 31;
    int t = wv >> 4;
    int r = wv & 15;
    const float* xr = x + (long)t * IN;
    const float* ar = loraA + (long)r * IN;
    float s = 0.0f;
    for (int i = lane; i < IN; i += 32) s += xr[i] * ar[i];
#pragma unroll
    for (int o = 16; o > 0; o >>= 1) s += __shfl_xor(s, o, 32);
    if (lane == 0) xa[t * RNK + r] = (_Float16)(2.0f * s);  // LORA_SCALING folded in
}

// ---------------- main: NF4-dequant GEMM + LoRA epilogue via WMMA ----------------
// Block = 256 threads (8 waves); wave w owns output columns [ (bx*8+w)*16 , +16 ),
// all 64 token rows (4 M-tiles), full K loop. Grid = 14336/128 = 112 blocks.
// Memory-bound on the 235 MB codes stream -> double-buffer codes in registers so
// the non-temporal HBM stream stays in flight across the WMMA sequence.
__global__ __launch_bounds__(256, 2) void nf4_lora_wmma(
    const int* __restrict__ codes, const float* __restrict__ absmax,
    const float* __restrict__ loraB, const _Float16* __restrict__ xh,
    const _Float16* __restrict__ xa, float* __restrict__ out) {
    const int lane  = threadIdx.x & 31;
    const int wave  = threadIdx.x >> 5;
    const int n0    = (blockIdx.x * 8 + wave) * 16;
    const int ncol  = lane & 15;       // B/D column, A row within tile
    const int hi16  = lane >> 4;
    const int khalfA = hi16 * 16;      // A-layout: upper half-wave holds K 16..31
    const int koffB  = hi16 * 16;      // B-layout: upper half-wave holds K 16..31
    const int n = n0 + ncol;           // output column owned by this lane

    const float tab = NF4_C[ncol];     // codebook resident in lanes 0..15 (dup 16..31)

    v8f acc[4] = {};

    const int*   crow  = codes  + (long)n * IN + koffB;
    const float* amrow = absmax + (long)n * (IN / NBLK);

    // ---- prologue of the software pipeline: fetch codes/absmax for kb = 0 ----
    v4i c[4];
    c[0] = __builtin_nontemporal_load((const v4i*)(crow + 0));
    c[1] = __builtin_nontemporal_load((const v4i*)(crow + 4));
    c[2] = __builtin_nontemporal_load((const v4i*)(crow + 8));
    c[3] = __builtin_nontemporal_load((const v4i*)(crow + 12));
    float am = amrow[koffB >> 6];

    for (int kb = 0; kb < IN; kb += 32) {
        // ---- dequant current codes -> B tile (32x16 f16) ----
        U16h b;   // lane = col, u[j] packs k = koffB+2j, +2j+1
        b.u[0] = pk2(nf4v(tab, c[0].x) * am, nf4v(tab, c[0].y) * am);
        b.u[1] = pk2(nf4v(tab, c[0].z) * am, nf4v(tab, c[0].w) * am);
        b.u[2] = pk2(nf4v(tab, c[1].x) * am, nf4v(tab, c[1].y) * am);
        b.u[3] = pk2(nf4v(tab, c[1].z) * am, nf4v(tab, c[1].w) * am);
        b.u[4] = pk2(nf4v(tab, c[2].x) * am, nf4v(tab, c[2].y) * am);
        b.u[5] = pk2(nf4v(tab, c[2].z) * am, nf4v(tab, c[2].w) * am);
        b.u[6] = pk2(nf4v(tab, c[3].x) * am, nf4v(tab, c[3].y) * am);
        b.u[7] = pk2(nf4v(tab, c[3].z) * am, nf4v(tab, c[3].w) * am);

        // ---- prefetch next K-step's codes while WMMAs run (keep HBM busy) ----
        if (kb + 32 < IN) {
            const int* cn = crow + kb + 32;
            c[0] = __builtin_nontemporal_load((const v4i*)(cn + 0));
            c[1] = __builtin_nontemporal_load((const v4i*)(cn + 4));
            c[2] = __builtin_nontemporal_load((const v4i*)(cn + 8));
            c[3] = __builtin_nontemporal_load((const v4i*)(cn + 12));
            am   = amrow[(kb + 32 + koffB) >> 6];
        }

        // ---- load all 4 A-tiles into distinct registers, then WMMA back-to-back ----
        const _Float16* ap = xh + (long)ncol * IN + kb + khalfA;
        U16h a0, a1, a2, a3;           // A tile 16x32 f16, element s = k 16*hi16+s (same as B)
        a0.q[0] = *(const v4i*)(ap);                 a0.q[1] = *(const v4i*)(ap + 8);
        a1.q[0] = *(const v4i*)(ap + 16 * IN);       a1.q[1] = *(const v4i*)(ap + 16 * IN + 8);
        a2.q[0] = *(const v4i*)(ap + 32 * IN);       a2.q[1] = *(const v4i*)(ap + 32 * IN + 8);
        a3.q[0] = *(const v4i*)(ap + 48 * IN);       a3.q[1] = *(const v4i*)(ap + 48 * IN + 8);

        acc[0] = __builtin_amdgcn_wmma_f32_16x16x32_f16(
            false, a0.h, false, b.h, (short)0, acc[0], false, false);
        acc[1] = __builtin_amdgcn_wmma_f32_16x16x32_f16(
            false, a1.h, false, b.h, (short)0, acc[1], false, false);
        acc[2] = __builtin_amdgcn_wmma_f32_16x16x32_f16(
            false, a2.h, false, b.h, (short)0, acc[2], false, false);
        acc[3] = __builtin_amdgcn_wmma_f32_16x16x32_f16(
            false, a3.h, false, b.h, (short)0, acc[3], false, false);
    }

    // ---- LoRA epilogue: one WMMA per M-tile with K = [r(16) | zeros(16)] ----
    U16h bl;
#pragma unroll
    for (int j = 0; j < 8; ++j) bl.u[j] = 0;
    if (hi16 == 0) {                       // lanes 0-15 carry K=0..15 -> r=0..15
        const float* bp = loraB + (long)n * RNK;
#pragma unroll
        for (int j = 0; j < 8; ++j) bl.u[j] = pk2(bp[2 * j], bp[2 * j + 1]);
    }                                       // lanes 16-31: K=16..31 -> zero pad

#pragma unroll
    for (int mt = 0; mt < 4; ++mt) {
        U16h al;                            // xa tile: lanes 0-15 elements 0..15 = r=0..15
        al.q[0] = hi16 ? (v4i)0 : *(const v4i*)(xa + (mt * 16 + ncol) * RNK);
        al.q[1] = hi16 ? (v4i)0 : *(const v4i*)(xa + (mt * 16 + ncol) * RNK + 8);   // K >= 16 zero pad
        acc[mt] = __builtin_amdgcn_wmma_f32_16x16x32_f16(
            false, al.h, false, bl.h, (short)0, acc[mt], false, false);
        // D layout: VGPR v -> row v (+8 for upper half-wave), col = lane&15
#pragma unroll
        for (int v = 0; v < 8; ++v)
            out[(long)(mt * 16 + hi16 * 8 + v) * OUTD + n0 + ncol] = acc[mt][v];
    }
}

extern "C" void kernel_launch(void* const* d_in, const int* in_sizes, int n_in,
                              void* d_out, int out_size, void* d_ws, size_t ws_size,
                              hipStream_t stream) {
    const float* x      = (const float*)d_in[0];
    const int*   codes  = (const int*)d_in[1];
    const float* absmax = (const float*)d_in[2];
    const float* loraA  = (const float*)d_in[3];
    const float* loraB  = (const float*)d_in[4];
    float*       out    = (float*)d_out;

    _Float16* xh = (_Float16*)d_ws;                                    // 64*4096 f16 = 512 KB
    _Float16* xa = (_Float16*)((char*)d_ws + (size_t)TOK * IN * 2);    // 64*16 f16  = 2 KB

    // prologue: f16 copy of x, and scaled LoRA activation xa = 2*x@A^T
    cvt_x_kernel<<<(TOK * IN / 4 + 255) / 256, 256, 0, stream>>>(x, xh);
    xa_kernel<<<(TOK * RNK * 32 + 255) / 256, 256, 0, stream>>>(x, loraA, xa);

    // main fused dequant-GEMM + LoRA: 112 blocks x 8 waves, each wave = 16 columns
    nf4_lora_wmma<<<OUTD / 128, 256, 0, stream>>>(codes, absmax, loraB, xh, xa, out);
}
